// ESACustomBotRGCN_32590211842593
// MI455X (gfx1250) — matrix operand order repaired and checked by reading, not compiled, not run on hardware
//
#include <hip/hip_runtime.h>
#include <hip/hip_bf16.h>
#include <stdint.h>

typedef __attribute__((ext_vector_type(16))) _Float16 v16h;
typedef __attribute__((ext_vector_type(8)))  float    v8f;

#define LRELU_SLOPE 0.01f
#define USE_ASYNC_STAGE 1

// ---------------------------------------------------------------------------
// B-fragment packing, k-major: Bh[kb][nt][lane][e]
//   k = kb*32 + 8*(lane>>4) + (e&7) + 16*(e>>3) ,  n = nt*16 + (lane&15)
// One k-chunk (KC kblks, all 8 ntiles) is contiguous -> single async LDS copy.
// ---------------------------------------------------------------------------
__device__ __forceinline__ float wenc_val(int k, int j,
                                          const float* wdes, const float* wtw,
                                          const float* wnum, const float* wcat,
                                          const float* wnew) {
  if (j < 28)  { return (k < 768)              ? wdes[j * 768 + k]               : 0.f; }
  if (j < 64)  { return (k >= 768 && k < 1536) ? wtw[(j - 28) * 768 + (k - 768)] : 0.f; }
  if (j < 76)  { return (k >= 1536 && k < 1543)? wnum[(j - 64) * 7 + (k - 1536)] : 0.f; }
  if (j < 116) { return (k >= 1543 && k < 1554)? wcat[(j - 76) * 11 + (k - 1543)]: 0.f; }
  return (k == 1554) ? wnew[j - 116] : 0.f;
}

__global__ void pack_w_encoder(const float* __restrict__ wdes, const float* __restrict__ wtw,
                               const float* __restrict__ wnum, const float* __restrict__ wcat,
                               const float* __restrict__ wnew, _Float16* __restrict__ Bh,
                               int Kblks) {
  int idx = blockIdx.x * blockDim.x + threadIdx.x;
  if (idx >= Kblks * 8 * 512) return;
  int e    = idx & 15;
  int lane = (idx >> 4) & 31;
  int nt   = (idx >> 9) & 7;
  int kb   = idx >> 12;
  int k = kb * 32 + 8 * (lane >> 4) + (e & 7) + 16 * (e >> 3);
  int n = nt * 16 + (lane & 15);
  Bh[idx] = (_Float16)wenc_val(k, n, wdes, wtw, wnum, wcat, wnew);
}

__global__ void pack_benc(const float* __restrict__ bd, const float* __restrict__ bt,
                          const float* __restrict__ bn, const float* __restrict__ bc,
                          const float* __restrict__ bnf, float* __restrict__ benc) {
  int j = threadIdx.x;
  float v;
  if      (j < 28)  v = bd[j];
  else if (j < 64)  v = bt[j - 28];
  else if (j < 76)  v = bn[j - 64];
  else if (j < 116) v = bc[j - 76];
  else              v = bnf[j - 116];
  benc[j] = v;
}

// 128x128 f32 weight -> k-major B fragments.  transpose=1: B(k,n)=W[n*128+k]
__global__ void pack_w128(const float* __restrict__ W, _Float16* __restrict__ Bh,
                          int transpose) {
  int idx = blockIdx.x * blockDim.x + threadIdx.x;   // 4*8*512 = 16384
  if (idx >= 4 * 8 * 512) return;
  int e    = idx & 15;
  int lane = (idx >> 4) & 31;
  int nt   = (idx >> 9) & 7;
  int kb   = idx >> 12;
  int k = kb * 32 + 8 * (lane >> 4) + (e & 7) + 16 * (e >> 3);
  int n = nt * 16 + (lane & 15);
  float v = transpose ? W[n * 128 + k] : W[k * 128 + n];
  Bh[idx] = (_Float16)v;
}

// Concatenate the 5 f32 inputs into one row-major f16 matrix [N, 1568]
__global__ void pack_xcat(const float* __restrict__ des, const float* __restrict__ tw,
                          const float* __restrict__ num, const float* __restrict__ cat,
                          const float* __restrict__ nf, _Float16* __restrict__ X, int N) {
  long long idx = (long long)blockIdx.x * blockDim.x + threadIdx.x;
  if (idx >= (long long)N * 1568) return;
  int n = (int)(idx / 1568);
  int k = (int)(idx % 1568);
  float v;
  if      (k < 768)   v = des[(size_t)n * 768 + k];
  else if (k < 1536)  v = tw[(size_t)n * 768 + (k - 768)];
  else if (k < 1543)  v = num[(size_t)n * 7 + (k - 1536)];
  else if (k < 1554)  v = cat[(size_t)n * 11 + (k - 1543)];
  else if (k == 1554) v = nf[n];
  else                v = 0.f;
  X[idx] = (_Float16)v;
}

// ---------------------------------------------------------------------------
// WMMA GEMM:  C[M,128] = act( A[M,lda] x B + bias ).  One wave produces a full
// 16x128 strip (8 accumulators) so A is read from HBM exactly once.  B staged
// chunk-wise into LDS via async-to-LDS; 8 waves/block share it.
// ---------------------------------------------------------------------------
template <int KBLKS, int KC>
__global__ void wmma_gemm128(const _Float16* __restrict__ A, int lda,
                             const _Float16* __restrict__ Bh,   // [KBLKS][8][512]
                             const float* __restrict__ bias,
                             float* __restrict__ Cf, _Float16* __restrict__ Ch,
                             int M, int relu) {
  __shared__ _Float16 ldsB[KC * 8 * 512];
  const int tid  = threadIdx.x;
  const int wave = tid >> 5;
  const int lane = tid & 31;
  const int mtile = blockIdx.x * 8 + wave;
  const int m    = lane & 15;
  const int half = lane >> 4;
  const int row0 = mtile * 16;
  int rowA = row0 + m; if (rowA > M - 1) rowA = M - 1;  // clamped rows discarded at store
  const _Float16* arow = A + (size_t)rowA * lda + 8 * half;

  v8f acc[8];
#pragma unroll
  for (int nt = 0; nt < 8; ++nt) acc[nt] = v8f{0.f,0.f,0.f,0.f,0.f,0.f,0.f,0.f};

  for (int kc0 = 0; kc0 < KBLKS; kc0 += KC) {
    __syncthreads();                                   // previous chunk consumed
    const char* gsrc = (const char*)(Bh + (size_t)kc0 * 8 * 512);
#if USE_ASYNC_STAGE
    {
      // low 32 bits of the flat shared address == logical LDS byte address
      uint32_t lbase = (uint32_t)(uintptr_t)(&ldsB[0]);
      for (int u = tid; u < KC * 512; u += 256) {
        uint64_t ga = (uint64_t)(uintptr_t)(gsrc + (size_t)u * 16);
        uint32_t la = lbase + (uint32_t)u * 16;
        asm volatile("global_load_async_to_lds_b128 %0, %1, off"
                     :: "v"(la), "v"(ga) : "memory");
      }
      asm volatile("s_wait_asynccnt 0x0" ::: "memory");
    }
#else
    {
      const uint32_t* s = (const uint32_t*)gsrc;
      uint32_t* d = (uint32_t*)ldsB;
      for (int u = tid; u < KC * 2048; u += 256) d[u] = s[u];
    }
#endif
    __syncthreads();

#pragma unroll
    for (int kb = 0; kb < KC; ++kb) {
      // A fragment (16-bit 16x32 layout): pairs k = 2p (p<4) / 2p+8 (p>=4), +8*half
      union { uint32_t u[8]; v16h v; } au;
      const uint32_t* ap = (const uint32_t*)(arow + (size_t)(kc0 + kb) * 32);
      au.u[0] = ap[0];  au.u[1] = ap[1];  au.u[2] = ap[2];  au.u[3] = ap[3];
      au.u[4] = ap[8];  au.u[5] = ap[9];  au.u[6] = ap[10]; au.u[7] = ap[11];
#pragma unroll
      for (int nt = 0; nt < 8; ++nt) {
        v16h b = *(const v16h*)(ldsB + ((size_t)(kb * 8 + nt) * 32 + lane) * 16);
        acc[nt] = __builtin_amdgcn_wmma_f32_16x16x32_f16(false, au.v, false, b,
                                                         (short)0, acc[nt], false, false);
      }
    }
  }

#pragma unroll
  for (int nt = 0; nt < 8; ++nt) {
    const int col = nt * 16 + m;                 // C/D: n = lane&15, m = r + 8*half
    const float bz = bias ? bias[col] : 0.f;
#pragma unroll
    for (int r = 0; r < 8; ++r) {
      int mrow = row0 + r + 8 * half;
      if (mrow < M) {
        float v = acc[nt][r] + bz;
        if (relu) v = (v >= 0.f) ? v : LRELU_SLOPE * v;
        size_t o = (size_t)mrow * 128 + col;
        if (Cf) Cf[o] = v;
        if (Ch) Ch[o] = (_Float16)v;
      }
    }
  }
}

// ---------------------------------------------------------------------------
// Edge phase: one wave32 per edge; float4/lane gather + atomic scatter.
// ---------------------------------------------------------------------------
__global__ void zero_f32(float* __restrict__ p, long long n) {
  long long i = (long long)blockIdx.x * blockDim.x + threadIdx.x;
  if (i < n) p[i] = 0.f;
}

__global__ void edge_agg(const int* __restrict__ ei, const int* __restrict__ et,
                         const float* __restrict__ t0, const float* __restrict__ t1,
                         float* __restrict__ agg0, float* __restrict__ agg1,
                         float* __restrict__ cnt, int E, int N) {
  int e = blockIdx.x * 8 + (threadIdx.x >> 5);
  int lane = threadIdx.x & 31;
  if (e >= E) return;
  int s = ei[e];
  int d = ei[E + e];
  int r = et[e];
  const float* t = (r == 0) ? t0 : t1;
  const float4 v = *(const float4*)(t + (size_t)s * 128 + lane * 4);
  float* ag = ((r == 0) ? agg0 : agg1) + (size_t)d * 128 + lane * 4;
  atomicAdd(ag + 0, v.x);
  atomicAdd(ag + 1, v.y);
  atomicAdd(ag + 2, v.z);
  atomicAdd(ag + 3, v.w);
  if (lane == 0) atomicAdd(cnt + (size_t)r * N + d, 1.0f);
}

// out = root + agg0/max(c0,1) + agg1/max(c1,1)  -> f16 activations for next GEMM
__global__ void combine_mean(const float* __restrict__ root,
                             const float* __restrict__ agg0, const float* __restrict__ agg1,
                             const float* __restrict__ cnt,
                             _Float16* __restrict__ xh, int N) {
  long long idx = (long long)blockIdx.x * blockDim.x + threadIdx.x;
  if (idx >= (long long)N * 128) return;
  int n = (int)(idx >> 7);
  float c0 = cnt[n];
  float c1 = cnt[N + n];
  float v = root[idx] + agg0[idx] / fmaxf(c0, 1.f) + agg1[idx] / fmaxf(c1, 1.f);
  xh[idx] = (_Float16)v;
}

// final tiny head: out[n,c] = b[c] + sum_k x2[n,k] * w2[c,k]   ([2,128])
__global__ void out_head(const float* __restrict__ x2, const float* __restrict__ w2,
                         const float* __restrict__ b2, float* __restrict__ out, int N) {
  int idx = blockIdx.x * blockDim.x + threadIdx.x;
  if (idx >= 2 * N) return;
  int n = idx >> 1, c = idx & 1;
  const float* xr = x2 + (size_t)n * 128;
  const float* wr = w2 + (size_t)c * 128;
  float s = b2[c];
#pragma unroll 4
  for (int k = 0; k < 128; ++k) s += xr[k] * wr[k];
  out[idx] = s;
}

// ---------------------------------------------------------------------------
extern "C" void kernel_launch(void* const* d_in, const int* in_sizes, int n_in,
                              void* d_out, int out_size, void* d_ws, size_t ws_size,
                              hipStream_t stream) {
  (void)n_in; (void)out_size; (void)ws_size;
  const int N = in_sizes[0] / 768;
  const int E = in_sizes[6];
  const int KB_ENC = 49;                 // 1568 / 32

  const float* des  = (const float*)d_in[0];
  const float* twt  = (const float*)d_in[1];
  const float* nump = (const float*)d_in[2];
  const float* catp = (const float*)d_in[3];
  const float* newf = (const float*)d_in[4];
  const int*   ei   = (const int*)d_in[5];   // JAX w/o x64: int64 request -> int32
  const int*   et   = (const int*)d_in[6];
  const float* w_des = (const float*)d_in[7],  * b_des = (const float*)d_in[8];
  const float* w_tw  = (const float*)d_in[9],  * b_tw  = (const float*)d_in[10];
  const float* w_num = (const float*)d_in[11], * b_num = (const float*)d_in[12];
  const float* w_cat = (const float*)d_in[13], * b_cat = (const float*)d_in[14];
  const float* w_new = (const float*)d_in[15], * b_new = (const float*)d_in[16];
  const float* w_in  = (const float*)d_in[17], * b_in  = (const float*)d_in[18];
  const float* r1_w  = (const float*)d_in[19], * r1_rt = (const float*)d_in[20], * r1_b = (const float*)d_in[21];
  const float* r2_w  = (const float*)d_in[22], * r2_rt = (const float*)d_in[23], * r2_b = (const float*)d_in[24];
  const float* w_o1  = (const float*)d_in[25], * b_o1  = (const float*)d_in[26];
  const float* w_o2  = (const float*)d_in[27], * b_o2  = (const float*)d_in[28];
  float* out = (float*)d_out;

  // ---- workspace layout (region0 aliased: xcat in phase 1, layer bufs later)
  char* ws = (char*)d_ws;
  const size_t S128 = (size_t)N * 128 * sizeof(float);
  const size_t xcat_bytes  = (size_t)N * 1568 * sizeof(_Float16);
  const size_t layer_bytes = 5 * S128 + (size_t)2 * N * sizeof(float);
  const size_t region0 = ((xcat_bytes > layer_bytes ? xcat_bytes : layer_bytes) + 255) & ~(size_t)255;

  _Float16* xcat = (_Float16*)ws;
  float*    root = (float*)ws;
  float*    t0   = (float*)(ws + 1 * S128);
  float*    t1   = (float*)(ws + 2 * S128);
  float*    agg0 = (float*)(ws + 3 * S128);
  float*    agg1 = (float*)(ws + 4 * S128);
  float*    cnt  = (float*)(ws + 5 * S128);
  size_t off = region0;
  _Float16* xh_a = (_Float16*)(ws + off); off += ((size_t)N * 128 * 2 + 255) & ~(size_t)255;
  _Float16* xh_b = (_Float16*)(ws + off); off += ((size_t)N * 128 * 2 + 255) & ~(size_t)255;
  _Float16* BhE  = (_Float16*)(ws + off); off += ((size_t)KB_ENC * 8 * 512 * 2 + 255) & ~(size_t)255;
  _Float16* BhW  = (_Float16*)(ws + off); off += ((size_t)4 * 8 * 512 * 2 + 255) & ~(size_t)255;
  float*    benc = (float*)(ws + off);

  const int Mtiles = (N + 15) / 16;
  dim3 ggrid((Mtiles + 7) / 8);

  // ---- phase 1: encoders + input linear (all WMMA) ----
  pack_w_encoder<<<(KB_ENC * 8 * 512 + 255) / 256, 256, 0, stream>>>(
      w_des, w_tw, w_num, w_cat, w_new, BhE, KB_ENC);
  pack_benc<<<1, 128, 0, stream>>>(b_des, b_tw, b_num, b_cat, b_new, benc);
  {
    long long tot = (long long)N * 1568;
    pack_xcat<<<(unsigned)((tot + 255) / 256), 256, 0, stream>>>(des, twt, nump, catp, newf, xcat, N);
  }
  wmma_gemm128<49, 7><<<ggrid, 256, 0, stream>>>(xcat, 1568, BhE, benc,
                                                 nullptr, xh_a, N, 1);
  pack_w128<<<64, 256, 0, stream>>>(w_in, BhW, 1);
  wmma_gemm128<4, 4><<<ggrid, 256, 0, stream>>>(xh_a, 128, BhW, b_in,
                                                nullptr, xh_b, N, 1);

  // ---- RGCN layers ----
  const long long nelem = (long long)N * 128;
  const unsigned gN128  = (unsigned)((nelem + 255) / 256);
  const unsigned gEdges = (unsigned)((E + 7) / 8);
  for (int layer = 0; layer < 2; ++layer) {
    const _Float16* xin = (layer == 0) ? xh_b : xh_a;
    _Float16*       xo  = (layer == 0) ? xh_a : xh_b;
    const float* wrel = (layer == 0) ? r1_w : r2_w;
    const float* wrt  = (layer == 0) ? r1_rt : r2_rt;
    const float* bb   = (layer == 0) ? r1_b : r2_b;

    pack_w128<<<64, 256, 0, stream>>>(wrt, BhW, 0);
    wmma_gemm128<4, 4><<<ggrid, 256, 0, stream>>>(xin, 128, BhW, bb, root, nullptr, N, 0);
    pack_w128<<<64, 256, 0, stream>>>(wrel, BhW, 0);
    wmma_gemm128<4, 4><<<ggrid, 256, 0, stream>>>(xin, 128, BhW, nullptr, t0, nullptr, N, 0);
    pack_w128<<<64, 256, 0, stream>>>(wrel + 128 * 128, BhW, 0);
    wmma_gemm128<4, 4><<<ggrid, 256, 0, stream>>>(xin, 128, BhW, nullptr, t1, nullptr, N, 0);

    zero_f32<<<gN128, 256, 0, stream>>>(agg0, nelem);
    zero_f32<<<gN128, 256, 0, stream>>>(agg1, nelem);
    zero_f32<<<(unsigned)((2LL * N + 255) / 256), 256, 0, stream>>>(cnt, 2LL * N);
    edge_agg<<<gEdges, 256, 0, stream>>>(ei, et, t0, t1, agg0, agg1, cnt, E, N);
    combine_mean<<<gN128, 256, 0, stream>>>(root, agg0, agg1, cnt, xo, N);
  }

  // ---- head ----
  pack_w128<<<64, 256, 0, stream>>>(w_o1, BhW, 1);
  wmma_gemm128<4, 4><<<ggrid, 256, 0, stream>>>(xh_b, 128, BhW, b_o1, t0, nullptr, N, 1);
  out_head<<<(unsigned)((2 * N + 255) / 256), 256, 0, stream>>>(t0, w_o2, b_o2, out, N);
}
